// GCN_23673859735659
// MI455X (gfx1250) — matrix operand order repaired and checked by reading, not compiled, not run on hardware
//
#include <hip/hip_runtime.h>
#include <hip/hip_bf16.h>

#define N_NODES 100000
#define N_EDGES 1600000
#define EMB     32
#define LDSW    34   // padded row stride (floats): even (b64-aligned) + conflict-free

typedef __attribute__((ext_vector_type(16))) __bf16 v16bf;
typedef __attribute__((ext_vector_type(8)))  __bf16 v8bf;
typedef __attribute__((ext_vector_type(8)))  float  v8f;
typedef __attribute__((ext_vector_type(2)))  float  v2f;

static __device__ __forceinline__ v16bf cat16(v8bf lo, v8bf hi) {
    return __builtin_shufflevector(lo, hi, 0,1,2,3,4,5,6,7,8,9,10,11,12,13,14,15);
}

// ---------------------------------------------------------------------------
// Kernel 0: prep. W2eT[n][k] = bf16(W2e[k][n])  (32x64 bf16)
//           W3p[k][n] = (k<35 && n<8) ? W3[k][n] : 0   (36x16 f32)
// ---------------------------------------------------------------------------
__global__ void prep_kernel(const float* __restrict__ W2e,
                            const float* __restrict__ W3,
                            __bf16* __restrict__ w2eT,
                            float*  __restrict__ w3p) {
    int idx = blockIdx.x * blockDim.x + threadIdx.x;   // 0..2047
    if (idx < 64 * 32) {
        int n = idx >> 6;   // 0..31
        int k = idx & 63;   // 0..63
        w2eT[idx] = (__bf16)W2e[k * 32 + n];
    }
    if (idx < 36 * 16) {
        int k = idx >> 4;   // 0..35
        int n = idx & 15;   // 0..15
        w3p[idx] = (k < 35 && n < 8) ? W3[k * 8 + n] : 0.f;
    }
}

// ---------------------------------------------------------------------------
// Kernel 1: per-node. emb gather -> f32 output + bf16 scratch; cars1; zero accs.
// ---------------------------------------------------------------------------
__global__ void node_kernel(const int* __restrict__ features,
                            const float* __restrict__ cars,
                            const float* __restrict__ embed_table,
                            const float* __restrict__ W2n,
                            const float* __restrict__ b2n,
                            float* __restrict__ emb_out,
                            __bf16* __restrict__ embb,
                            float* __restrict__ cars1,
                            float* __restrict__ cars_out,
                            float* __restrict__ entered_out,
                            int*   __restrict__ mmax,
                            float* __restrict__ ssum) {
    int i = blockIdx.x * blockDim.x + threadIdx.x;
    if (i >= N_NODES) return;
    int f = features[i];
    const float* row = embed_table + (size_t)f * EMB;
    float e[EMB];
    float acc = b2n[0];
    #pragma unroll
    for (int t = 0; t < EMB; ++t) { e[t] = row[t]; acc += e[t] * W2n[t]; }
    float c = cars[i];
    acc += c * W2n[EMB];                       // parked
    float cars1v = fmaxf(fmaxf(acc, 0.f) + c, 0.f);
    #pragma unroll
    for (int t = 0; t < EMB; ++t) {
        emb_out[(size_t)i * EMB + t] = e[t];
        embb[(size_t)i * EMB + t]    = (__bf16)e[t];
    }
    cars1[i]       = cars1v;
    cars_out[i]    = 0.f;
    entered_out[i] = 0.f;
    mmax[i]        = 0;      // logits >= 0, so int-0 is a valid float-max identity
    ssum[i]        = 0.f;
}

// ---------------------------------------------------------------------------
// Kernel 2: per-edge. Stage 1: he = [emb_s|emb_d] @ W2e  (4x wmma bf16, K=64).
//           Stage 2: h3 = relu(X(16x36) @ W3p + b3)      (9x wmma f32, K=4).
//           Tail: logit = relu(h3 @ W4 + b4) * enabled.
// 256 threads = 8 waves, one 16-edge tile per wave.
// ---------------------------------------------------------------------------
__global__ __launch_bounds__(256) void edge_kernel(
        const int*   __restrict__ src,
        const int*   __restrict__ dst,
        const __bf16* __restrict__ embb,
        const __bf16* __restrict__ w2eT,
        const float* __restrict__ w3p,
        const float* __restrict__ cars1,
        const float* __restrict__ entered,
        const int*   __restrict__ features,
        const float* __restrict__ freev,
        const float* __restrict__ b2e,
        const float* __restrict__ b3,
        const float* __restrict__ W4,
        const float* __restrict__ b4,
        float* __restrict__ logit) {
    __shared__ float lds_he[8][16][LDSW];   // ~17 KB; he tile, later reused for h3

    int lane = threadIdx.x & 31;
    int wave = threadIdx.x >> 5;
    int m    = lane & 15;                  // A-row / C-column index
    int half = lane >> 4;                  // selects K slice / C-row block
    long base = ((long)blockIdx.x * 8 + wave) * 16;
    long e = base + m;
    if (e >= N_EDGES) e = N_EDGES - 1;     // clamp; all lanes stay active for WMMA
    int s = src[e], d = dst[e];

    // ---- Stage 1: A fragments (16x32 bf16): two b128 loads per lane --------
    const __bf16* ps = embb + (size_t)s * EMB;
    const __bf16* pd = embb + (size_t)d * EMB;
    v16bf a_src = cat16(*(const v8bf*)(ps + half * 8), *(const v8bf*)(ps + 16 + half * 8));
    v16bf a_dst = cat16(*(const v8bf*)(pd + half * 8), *(const v8bf*)(pd + 16 + half * 8));

    // B fragments (32x16 bf16): lane holds col n=(lane&15)+n0, K = kbase+8*half+j (j<8), kbase+16+8*half+(j-8) (j>=8)
    auto ldB = [&](int n0, int kbase) -> v16bf {
        const __bf16* p = w2eT + (size_t)(m + n0) * 64 + kbase + 8 * half;
        return cat16(*(const v8bf*)p, *(const v8bf*)(p + 16));
    };
    v16bf b_t0 = ldB(0, 0),  b_b0 = ldB(0, 32);
    v16bf b_t1 = ldB(16, 0), b_b1 = ldB(16, 32);

    v8f c0 = {}, c1 = {};
    c0 = __builtin_amdgcn_wmma_f32_16x16x32_bf16(false, a_src, false, b_t0, (short)0, c0, false, false);
    c0 = __builtin_amdgcn_wmma_f32_16x16x32_bf16(false, a_dst, false, b_b0, (short)0, c0, false, false);
    c1 = __builtin_amdgcn_wmma_f32_16x16x32_bf16(false, a_src, false, b_t1, (short)0, c1, false, false);
    c1 = __builtin_amdgcn_wmma_f32_16x16x32_bf16(false, a_dst, false, b_b1, (short)0, c1, false, false);

    // Spill he (+ b2e folded in, per-column constant) to LDS in D layout.
    float b2e_lo = b2e[m], b2e_hi = b2e[16 + m];
    #pragma unroll
    for (int r = 0; r < 8; ++r) {
        lds_he[wave][r + 8 * half][m]      = c0[r] + b2e_lo;
        lds_he[wave][r + 8 * half][16 + m] = c1[r] + b2e_hi;
    }
    __syncthreads();

    // ---- Stage 2: X(16x36) @ W3p(36x16) via 9x V_WMMA_F32_16X16X4_F32 ------
    // X row m: [he[0..31], cars1[s], cars1[d], entered[s], 0]
    float cs = cars1[s], cd = cars1[d], en = entered[s];
    const float* Xrow = &lds_he[wave][m][0];
    v8f h = {};
    #pragma unroll
    for (int bks = 0; bks < 8; ++bks) {
        int k = 4 * bks + 2 * half;                 // A: lane holds K = 2*half + j
        v2f a;  a[0] = Xrow[k];          a[1] = Xrow[k + 1];
        v2f bb; bb[0] = w3p[k * 16 + m]; bb[1] = w3p[(k + 1) * 16 + m];
        h = __builtin_amdgcn_wmma_f32_16x16x4_f32(false, a, false, bb, (short)0, h, false, false);
    }
    {   // K block 32..35 from registers (row 35 of W3p is zero anyway)
        int k = 32 + 2 * half;
        v2f a;  a[0] = half ? en : cs;   a[1] = half ? 0.f : cd;
        v2f bb; bb[0] = w3p[k * 16 + m]; bb[1] = w3p[(k + 1) * 16 + m];
        h = __builtin_amdgcn_wmma_f32_16x16x4_f32(false, a, false, bb, (short)0, h, false, false);
    }

    // h3 = relu(h + b3[n]); store back into LDS (cols 0..15), D layout.
    float b3n = (m < 8) ? b3[m] : 0.f;
    #pragma unroll
    for (int r = 0; r < 8; ++r)
        lds_he[wave][r + 8 * half][m] = fmaxf(h[r] + b3n, 0.f);

    __syncthreads();

    // ---- Tail: h4 = h3 @ W4 + b4 (8 FMAs/edge), logit mask ------------------
    if (half == 0 && base + m < N_EDGES) {
        float acc4 = b4[0];
        #pragma unroll
        for (int q = 0; q < 8; ++q)
            acc4 += lds_he[wave][m][q] * W4[q];
        float selfloop = (features[s] == features[d]) ? 1.f : 0.f;
        float enabled  = (selfloop != freev[s]) ? 1.f : 0.f;
        logit[base + m] = fmaxf(acc4, 0.f) * enabled;
    }
}

// ---------------------------------------------------------------------------
// Kernel 3: segment max over src (logits >= 0 -> int atomicMax on bits).
// ---------------------------------------------------------------------------
__global__ void smax_kernel(const int* __restrict__ src,
                            const float* __restrict__ logit,
                            int* __restrict__ mmax) {
    long e = (long)blockIdx.x * blockDim.x + threadIdx.x;
    if (e >= N_EDGES) return;
    atomicMax(&mmax[src[e]], __float_as_int(logit[e]));
}

// ---------------------------------------------------------------------------
// Kernel 4: ex = exp(l - m[src]); ssum = segment_sum(ex, src)
// ---------------------------------------------------------------------------
__global__ void ssum_kernel(const int* __restrict__ src,
                            const float* __restrict__ logit,
                            const int* __restrict__ mmax,
                            float* __restrict__ ex,
                            float* __restrict__ ssum) {
    long e = (long)blockIdx.x * blockDim.x + threadIdx.x;
    if (e >= N_EDGES) return;
    int s = src[e];
    float x = __expf(logit[e] - __int_as_float(mmax[s]));
    ex[e] = x;
    atomicAdd(&ssum[s], x);
}

// ---------------------------------------------------------------------------
// Kernel 5: prob, scatter c / o into cars_out / entered_out by dst.
// ---------------------------------------------------------------------------
__global__ void scatter_kernel(const int* __restrict__ src,
                               const int* __restrict__ dst,
                               const int* __restrict__ features,
                               const float* __restrict__ ex,
                               const float* __restrict__ ssum,
                               const float* __restrict__ cars1,
                               float* __restrict__ cars_out,
                               float* __restrict__ entered_out) {
    long e = (long)blockIdx.x * blockDim.x + threadIdx.x;
    if (e >= N_EDGES) return;
    int s = src[e], d = dst[e];
    float prob = ex[e] / ssum[s];
    float c = cars1[s] * prob;
    float selfloop = (features[s] == features[d]) ? 1.f : 0.f;
    atomicAdd(&cars_out[d], c);
    atomicAdd(&entered_out[d], c * (1.f - selfloop));
}

// ---------------------------------------------------------------------------
extern "C" void kernel_launch(void* const* d_in, const int* in_sizes, int n_in,
                              void* d_out, int out_size, void* d_ws, size_t ws_size,
                              hipStream_t stream) {
    const int*   features    = (const int*)d_in[0];
    const float* cars        = (const float*)d_in[1];
    const float* entered     = (const float*)d_in[2];
    const float* freev       = (const float*)d_in[3];
    const int*   src         = (const int*)d_in[4];
    const int*   dst         = (const int*)d_in[5];
    const float* embed_table = (const float*)d_in[6];
    const float* W2n         = (const float*)d_in[7];
    const float* b2n         = (const float*)d_in[8];
    const float* W2e         = (const float*)d_in[9];
    const float* b2e         = (const float*)d_in[10];
    const float* W3          = (const float*)d_in[11];
    const float* b3          = (const float*)d_in[12];
    const float* W4          = (const float*)d_in[13];
    const float* b4          = (const float*)d_in[14];

    // Output layout: cars_out[N], emb[N*32], entered_out[N]
    float* out         = (float*)d_out;
    float* cars_out    = out;
    float* emb_out     = out + N_NODES;
    float* entered_out = out + N_NODES + (size_t)N_NODES * EMB;

    // Workspace carve-up (256B aligned slices)
    char* ws = (char*)d_ws;
    size_t off = 0;
    auto wsalloc = [&](size_t bytes) -> void* {
        void* p = ws + off;
        off += (bytes + 255) & ~(size_t)255;
        return p;
    };
    __bf16* embb  = (__bf16*)wsalloc((size_t)N_NODES * EMB * sizeof(__bf16)); // 6.4 MB
    float*  cars1 = (float*)wsalloc((size_t)N_NODES * sizeof(float));         // 0.4 MB
    __bf16* w2eT  = (__bf16*)wsalloc(64 * 32 * sizeof(__bf16));               // 4 KB
    float*  w3p   = (float*)wsalloc(36 * 16 * sizeof(float));                 // 2.3 KB
    float*  logit = (float*)wsalloc((size_t)N_EDGES * sizeof(float));         // 6.4 MB
    float*  exbuf = (float*)wsalloc((size_t)N_EDGES * sizeof(float));         // 6.4 MB
    int*    mmax  = (int*)wsalloc((size_t)N_NODES * sizeof(int));             // 0.4 MB
    float*  ssum  = (float*)wsalloc((size_t)N_NODES * sizeof(float));         // 0.4 MB
    (void)ws_size;

    prep_kernel<<<8, 256, 0, stream>>>(W2e, W3, w2eT, w3p);

    node_kernel<<<(N_NODES + 255) / 256, 256, 0, stream>>>(
        features, cars, embed_table, W2n, b2n,
        emb_out, embb, cars1, cars_out, entered_out, mmax, ssum);

    // 8 waves/block * 16 edges/wave = 128 edges/block; 1.6M/128 = 12500 blocks
    edge_kernel<<<(N_EDGES + 127) / 128, 256, 0, stream>>>(
        src, dst, embb, w2eT, w3p, cars1, entered, features, freev,
        b2e, b3, W4, b4, logit);

    smax_kernel<<<(N_EDGES + 255) / 256, 256, 0, stream>>>(src, logit, mmax);

    ssum_kernel<<<(N_EDGES + 255) / 256, 256, 0, stream>>>(src, logit, mmax, exbuf, ssum);

    scatter_kernel<<<(N_EDGES + 255) / 256, 256, 0, stream>>>(
        src, dst, features, exbuf, ssum, cars1, cars_out, entered_out);
}
